// CBL_19533511262658
// MI455X (gfx1250) — compile-verified, matrix operand order checked by reading, not run on hardware
//
#include <hip/hip_runtime.h>
#include <cstdint>

// ---------------------------------------------------------------------------
// CDNA5 (gfx1250) context-loss kernel.
//   Pass 1: per-batch counts (boundary / selected).
//   Pass 2: er_input [B,C,H,W] f32 -> feat [B,H,W,C] f16 (151 MB, L2-resident)
//           + exact fp32 norms nf[B,H,W].
//   Pass 3: per 16-pixel strip, Gram bands vs 5 neighbor rows using
//           v_wmma_f32_16x16x32_f16, double-buffered B-fragment loads so the
//           matrix pipe never waits on L2; band extraction via LDS;
//           one float atomic per wave.
//   Pass 4: finalize scalar with the reference's exact normalization.
// ---------------------------------------------------------------------------

typedef _Float16 v16h  __attribute__((ext_vector_type(16)));
typedef _Float16 half8 __attribute__((ext_vector_type(8)));
typedef float    v8f   __attribute__((ext_vector_type(8)));

constexpr int Bn = 4, Cn = 128, Hn = 384, Wn = 384;
constexpr size_t HWn = (size_t)Hn * Wn;

__device__ __forceinline__ half8 ldh8(const _Float16* p) {
  return *reinterpret_cast<const half8*>(p);
}

struct BFrag { v16h f[4]; };

__device__ __forceinline__ void loadFrag(BFrag& bf, const _Float16* bp) {
#pragma unroll
  for (int kk = 0; kk < 4; ++kk) {
    union { v16h v; half8 h[2]; } u;
    u.h[0] = ldh8(bp + kk * 32);
    u.h[1] = ldh8(bp + kk * 32 + 16);
    bf.f[kk] = u.v;
  }
}

// ---------------------------------------------------------------------------
__global__ void ctx_init(float* lossSum, int* selCnt, int* bndCnt) {
  int t = threadIdx.x;
  if (t < Bn) { lossSum[t] = 0.f; selCnt[t] = 0; bndCnt[t] = 0; }
}

// ---------------------------------------------------------------------------
__global__ void ctx_count(const int* __restrict__ gt, const int* __restrict__ seg,
                          int* __restrict__ bndCnt, int* __restrict__ selCnt) {
  size_t tid = (size_t)blockIdx.x * blockDim.x + threadIdx.x;
  if (tid >= (size_t)Bn * HWn) return;
  int b = (int)(tid / HWn);
  size_t r = tid % HWn;
  int y = (int)(r / Wn), x = (int)(r % Wn);
  int gbv = gt[tid];                            if (gbv == 255) gbv = 0;
  int s1  = seg[((size_t)b * 2 + 1) * HWn + r]; if (s1 == 255) s1 = 0;
  bool bnd = (gbv * s1) > 0;
  bool sel = bnd && y >= 2 && y <= Hn - 3 && x >= 2 && x <= Wn - 3;
  unsigned long long mb = __ballot(bnd);
  unsigned long long ms = __ballot(sel);
  if ((threadIdx.x & 31) == 0) {                // waves never straddle a batch
    if (mb) atomicAdd(&bndCnt[b], (int)__popcll(mb));
    if (ms) atomicAdd(&selCnt[b], (int)__popcll(ms));
  }
}

// ---------------------------------------------------------------------------
// [B,C,H,W] f32 -> [B,H,W,C] f16 via conflict-free LDS transpose + fp32 norms.
__global__ void ctx_convert(const float* __restrict__ er,
                            _Float16* __restrict__ feat,
                            float* __restrict__ nf) {
  __shared__ _Float16 sh[64 * 130];             // 65-dword row stride: conflict-free
  __shared__ float nsh[256];
  const int t = threadIdx.x;
  const int tx = t & 63, cy = t >> 6;
  const int x0 = blockIdx.x * 64, y = blockIdx.y, b = blockIdx.z;
  const float* src = er + (size_t)b * Cn * HWn + (size_t)y * Wn + (x0 + tx);
  float nacc = 0.f;
  for (int cc = cy; cc < Cn; cc += 4) {         // coalesced per-plane reads
    float v = src[(size_t)cc * HWn];
    nacc += v * v;
    sh[tx * 130 + cc] = (_Float16)v;
  }
  nsh[t] = nacc;
  __syncthreads();
  if (t < 64) {
    float s = nsh[tx] + nsh[64 + tx] + nsh[128 + tx] + nsh[192 + tx];
    nf[((size_t)b * Hn + y) * Wn + x0 + tx] = sqrtf(s);
  }
  uint32_t* dst = reinterpret_cast<uint32_t*>(feat) +
                  (((size_t)b * Hn + y) * Wn + x0) * (Cn / 2);
  const uint32_t* shu = reinterpret_cast<const uint32_t*>(sh);
  for (int it = 0; it < 16; ++it) {             // contiguous coalesced store
    int g = it * 256 + t;
    int p = g >> 6, dw = g & 63;
    dst[g] = shu[p * 65 + dw];
  }
}

// ---------------------------------------------------------------------------
// One wave32 per (16-pixel strip, interior row, batch).
__global__ void __launch_bounds__(32)
ctx_main(const _Float16* __restrict__ feat, const float* __restrict__ nf,
         const int* __restrict__ gt, const int* __restrict__ seg,
         float* __restrict__ lossSum) {
  __shared__ float ldsD[16 * 36];               // [M=16][N=0..31], stride 36
  const int lane = threadIdx.x;
  const int m16  = lane & 15;
  const int lsel = (lane >> 4) * 8;             // K-half offset per ISA fragment layout
  const int x0 = blockIdx.x * 16;
  const int y  = 2 + blockIdx.y;
  const int b  = blockIdx.z;

  // A fragments: 16 pixels (y, x0..x0+15) x K=128, reused across all 5 dy.
  const _Float16* ap = feat + (((size_t)b * Hn + y) * Wn + (x0 + m16)) * Cn + lsel;
  v16h afrag[4];
#pragma unroll
  for (int kk = 0; kk < 4; ++kk) {
    union { v16h v; half8 h[2]; } u;
    u.h[0] = ldh8(ap + kk * 32);
    u.h[1] = ldh8(ap + kk * 32 + 16);
    afrag[kk] = u.v;
  }

  // B-fragment pointer for (dyi, tile): window col base x0-2 + 16*tile.
  auto bptr = [&](int dyi2, int tile) -> const _Float16* {
    int yq2 = y + dyi2 - 2;                     // in [0,383] for interior y
    int xq = x0 - 2 + tile * 16 + m16;
    xq = xq < 0 ? 0 : (xq > Wn - 1 ? Wn - 1 : xq);  // clamped lanes masked later
    return feat + (((size_t)b * Hn + yq2) * Wn + xq) * Cn + lsel;
  };

  // Per-pixel (lanes 0..15) state.
  const int xp = x0 + m16;
  const size_t pidx = ((size_t)b * Hn + y) * Wn + xp;
  int gbv = gt[pidx];                           if (gbv == 255) gbv = 0;
  int s0r = seg[((size_t)b * 2 + 0) * HWn + (size_t)y * Wn + xp];
  int s1r = seg[((size_t)b * 2 + 1) * HWn + (size_t)y * Wn + xp];
  int s1c = (s1r == 255) ? 0 : s1r;
  const bool selP = (lane < 16) && (gbv * s1c > 0) && (xp >= 2) && (xp <= Wn - 3);
  const float nfP = fmaxf(nf[pidx], 1e-8f);
  const float s0p = (float)s0r, s1p = (float)s1r;

  float lacc = 0.f;

  // Double-buffered B fragments: loads for dy+1 fly while dy's WMMAs +
  // extraction execute, so v_wmma never waits on a just-issued L2 load.
  BFrag eb0, eb1, ob0, ob1;                     // even/odd iteration buffers
  loadFrag(eb0, bptr(0, 0));
  loadFrag(eb1, bptr(0, 1));

#pragma unroll
  for (int dyi = 0; dyi < 5; ++dyi) {
    const int dy = dyi - 2;
    const int yq = y + dy;
    BFrag& c0 = (dyi & 1) ? ob0 : eb0;
    BFrag& c1 = (dyi & 1) ? ob1 : eb1;
    BFrag& n0 = (dyi & 1) ? eb0 : ob0;
    BFrag& n1 = (dyi & 1) ? eb1 : ob1;
    if (dyi < 4) {                              // prefetch next row's fragments
      loadFrag(n0, bptr(dyi + 1, 0));
      loadFrag(n1, bptr(dyi + 1, 1));
    }

    v8f acc0 = {}; v8f acc1 = {};
#pragma unroll
    for (int kk = 0; kk < 4; ++kk) {            // K = 128 in 4 steps of 32
      acc0 = __builtin_amdgcn_wmma_f32_16x16x32_f16(false, afrag[kk], false, c0.f[kk],
                                                    (short)0, acc0, false, false);
      acc1 = __builtin_amdgcn_wmma_f32_16x16x32_f16(false, afrag[kk], false, c1.f[kk],
                                                    (short)0, acc1, false, false);
    }

    // D layout: VGPR r, lanes 0-15 -> (M=r, N=lane); lanes 16-31 -> (M=8+r, N=lane-16).
    const int mrow = (lane >> 4) * 8;
#pragma unroll
    for (int r = 0; r < 8; ++r) {
      ldsD[(mrow + r) * 36 + m16]      = acc0[r];   // window cols 0..15
      ldsD[(mrow + r) * 36 + 16 + m16] = acc1[r];   // window cols 16..31
    }
    __syncthreads();

    if (selP) {                                 // lane m owns pixel x0+m; n = m+dx+2
#pragma unroll
      for (int dx = -2; dx <= 2; ++dx) {
        if (dy == 0 && dx == 0) continue;
        const float dotv = ldsD[lane * 36 + (lane + dx + 2)];
        const int xq = xp + dx;
        const size_t qpix = ((size_t)b * Hn + yq) * Wn + xq;
        const float nfQ = fmaxf(nf[qpix], 1e-8f);
        const float s0q = (float)seg[((size_t)b * 2 + 0) * HWn + (size_t)yq * Wn + xq];
        const float s1q = (float)seg[((size_t)b * 2 + 1) * HWn + (size_t)yq * Wn + xq];
        const float lab = s0p * s0q + s1p * s1q;
        const float cs = dotv / (nfP * nfQ);
        const float e = cs - lab;
        lacc += e * e;
      }
    }
    __syncthreads();
  }

#pragma unroll
  for (int off = 16; off > 0; off >>= 1)
    lacc += __shfl_xor(lacc, off, 32);
  if (lane == 0) atomicAdd(&lossSum[b], lacc);
}

// ---------------------------------------------------------------------------
__global__ void ctx_final(const float* lossSum, const int* selCnt,
                          const int* bndCnt, float* out) {
  float tot = 0.f; int nv = 0;
  for (int b = 0; b < Bn; ++b) {
    if (bndCnt[b] >= 1) {
      nv++;
      tot += lossSum[b] / fmaxf((float)selCnt[b], 1.f) * (1.f / 24.f);
    }
  }
  tot /= (float)(nv > 0 ? nv : 1);
  if (tot != tot) tot = 0.f;                    // NaN -> 0
  out[0] = tot;
}

// ---------------------------------------------------------------------------
extern "C" void kernel_launch(void* const* d_in, const int* in_sizes, int n_in,
                              void* d_out, int out_size, void* d_ws, size_t ws_size,
                              hipStream_t stream) {
  const float* er  = (const float*)d_in[0];     // [B,C,H,W] f32
  const int*   seg = (const int*)d_in[1];       // [B,2,H,W] i32
  const int*   gt  = (const int*)d_in[2];       // [B,H,W]   i32

  char* ws = (char*)d_ws;
  const size_t featB = (size_t)Bn * HWn * Cn * sizeof(_Float16); // ~151.0 MB
  const size_t nfB   = (size_t)Bn * HWn * sizeof(float);         // ~2.36 MB
  _Float16* feat  = (_Float16*)ws;
  float* nfp      = (float*)(ws + featB);
  float* lossSum  = (float*)(ws + featB + nfB);
  int*   selCnt   = (int*)  (ws + featB + nfB + 64);
  int*   bndCnt   = (int*)  (ws + featB + nfB + 128);

  ctx_init<<<1, 32, 0, stream>>>(lossSum, selCnt, bndCnt);
  ctx_count<<<(unsigned)((Bn * HWn) / 256), 256, 0, stream>>>(gt, seg, bndCnt, selCnt);
  ctx_convert<<<dim3(Wn / 64, Hn, Bn), 256, 0, stream>>>(er, feat, nfp);
  ctx_main<<<dim3(Wn / 16, Hn - 4, Bn), 32, 0, stream>>>(feat, nfp, gt, seg, lossSum);
  ctx_final<<<1, 1, 0, stream>>>(lossSum, selCnt, bndCnt, (float*)d_out);
}
